// TriangleMultiplication_82652350644901
// MI455X (gfx1250) — matrix operand order repaired and checked by reading, not compiled, not run on hardware
//
#include <hip/hip_runtime.h>
#include <stdint.h>

#define LDIM 512
#define CDIM 128
#define NROWS (LDIM * LDIM) // 262144

typedef __attribute__((ext_vector_type(16))) __bf16 bf16x16;
typedef __attribute__((ext_vector_type(8)))  __bf16 bf16x8;
typedef __attribute__((ext_vector_type(8)))  float  f32x8;
typedef __attribute__((ext_vector_type(4)))  float  f32x4;
typedef __attribute__((ext_vector_type(4)))  uint32_t u32x4;
typedef __attribute__((ext_vector_type(8)))  int    i32x8;
typedef __attribute__((ext_vector_type(4)))  int    i32x4;
typedef __attribute__((address_space(3)))    char   lds_char;

__device__ __forceinline__ __bf16 f2bf(float f) {
  union { float f; uint32_t u; } v; v.f = f;
  uint32_t r = v.u + 0x7FFFu + ((v.u >> 16) & 1u); // round-to-nearest-even
  uint16_t s = (uint16_t)(r >> 16);
  __bf16 o; __builtin_memcpy(&o, &s, 2); return o;
}
__device__ __forceinline__ float bf2f(__bf16 b) {
  uint16_t s; __builtin_memcpy(&s, &b, 2);
  union { uint32_t u; float f; } v; v.u = ((uint32_t)s) << 16;
  return v.f;
}
__device__ __forceinline__ float sigmoidf_(float x) {
  return 1.0f / (1.0f + __expf(-x));
}

#define BF16X16_CAT(lo, hi) \
  __builtin_shufflevector(lo, hi, 0,1,2,3,4,5,6,7,8,9,10,11,12,13,14,15)

#define WMMA_BF16(a, b, c) \
  __builtin_amdgcn_wmma_f32_16x16x32_bf16(false, a, false, b, (short)0, c, false, false)

// TDM: copy `nbytes/2` bf16 elements laid out as a (16384 x rows) 2-D tile
// from global `src` into LDS at byte offset `lds_off` (ISA ch.8 D# groups).
__device__ __forceinline__ void tdm_load_weights(const void* src, uint32_t lds_off,
                                                 uint32_t rows) {
  uint64_t ga = (uint64_t)(uintptr_t)src;
  u32x4 g0;
  g0[0] = 1u;                                   // count=1, user mode, no gather
  g0[1] = lds_off;                              // lds_addr (bytes)
  g0[2] = (uint32_t)ga;                         // global_addr[31:0]
  g0[3] = (uint32_t)(ga >> 32) | (2u << 30);    // global_addr[56:32] | type=2
  i32x8 g1;
  g1[0] = (int)(1u << 16);                      // wg_mask=0, data_size=1 (2B)
  g1[1] = (int)(16384u << 16);                  // tensor_dim0 = 16384 (lo16 in hi half)
  g1[2] = (int)(rows << 16);                    // tensor_dim0 hi=0 | tensor_dim1=rows
  g1[3] = (int)(16384u << 16);                  // tensor_dim1 hi=0 | tile_dim0=16384
  g1[4] = (int)rows;                            // tile_dim1=rows, tile_dim2=0
  g1[5] = 16384;                                // tensor_dim0_stride lo
  g1[6] = 0;                                    // stride0 hi | stride1 lo
  g1[7] = 0;
  i32x4 g2 = {0, 0, 0, 0};
  i32x4 g3 = {0, 0, 0, 0};
#if __has_include(<hip/amd_detail/amd_gfx1250_TDM.h>)
  i32x8 g4 = {0, 0, 0, 0, 0, 0, 0, 0};
  __builtin_amdgcn_tensor_load_to_lds(g0, g1, g2, g3, g4, 0);
#else
  __builtin_amdgcn_tensor_load_to_lds(g0, g1, g2, g3, 0);
#endif
}

// ---------------------------------------------------------------------------
// Kernel 1: transpose+convert 6 weight matrices (128x128 f32, [k][n]) into
// bf16 packed layout [mat][n][k] so WMMA B fragments are contiguous loads.
// mats: 0=left_w 1=right_w 2=lgate_w 3=rgate_w 4=fgate_w 5=out_w
// ---------------------------------------------------------------------------
__global__ void pack_weights(const float* __restrict__ lw, const float* __restrict__ rw,
                             const float* __restrict__ lgw, const float* __restrict__ rgw,
                             const float* __restrict__ fgw, const float* __restrict__ ow,
                             __bf16* __restrict__ wt) {
  int idx = blockIdx.x * blockDim.x + threadIdx.x;
  if (idx >= 6 * CDIM * CDIM) return;
  int m = idx / (CDIM * CDIM);
  int rem = idx % (CDIM * CDIM);
  int n = rem / CDIM, k = rem % CDIM;
  const float* src;
  switch (m) {
    case 0: src = lw; break;  case 1: src = rw; break;
    case 2: src = lgw; break; case 3: src = rgw; break;
    case 4: src = fgw; break; default: src = ow; break;
  }
  wt[idx] = f2bf(src[k * CDIM + n]);
}

// ---------------------------------------------------------------------------
// Kernel 2: fused LayerNorm + 5 projections (WMMA bf16) + masks/gates.
// Block = 256 threads (8 waves), 256 rows. The 5 packed weight matrices
// (160KB) are staged into LDS once per block by the Tensor Data Mover,
// overlapped with the LayerNorm phase. Outputs:
//   left_t / right_t : bf16 channel-major [c][row]  (for per-channel einsum)
//   gbuf             : f32 row-major [row][c]       (final sigmoid gate)
// ---------------------------------------------------------------------------
__global__ void stage1(const float* __restrict__ act, const float* __restrict__ mask,
                       const float* __restrict__ ng, const float* __restrict__ nb,
                       const float* __restrict__ lb, const float* __restrict__ rb,
                       const float* __restrict__ lgb, const float* __restrict__ rgb,
                       const float* __restrict__ fgb,
                       const __bf16* __restrict__ wt,
                       __bf16* __restrict__ left_t, __bf16* __restrict__ right_t,
                       float* __restrict__ gbuf) {
  extern __shared__ char smem[];
  __bf16* a_lds = (__bf16*)smem;                              // 64KB
  float*  m_lds = (float*)(smem + 256 * CDIM * 2);            // 1KB
  __bf16* w_lds = (__bf16*)(smem + 256 * CDIM * 2 + 256 * 4); // 160KB (5 mats)

  const int t = threadIdx.x;
  const int wave = t >> 5, lane = t & 31;
  const int n16 = lane & 15, half = lane >> 4;
  const int r0 = blockIdx.x * 256;

  // ---- Kick off TDM copy of the 5 projection matrices into LDS ----
  if (wave == 0) {
    uint32_t lds_off = (uint32_t)(uintptr_t)(lds_char*)(void*)w_lds;
    tdm_load_weights(wt, lds_off, 5u);
  }

  // ---- Phase 1 (overlapped with TDM): per-row LayerNorm over C ----
  {
    const float* row = act + (size_t)(r0 + t) * CDIM;
    float s = 0.f, ss = 0.f;
    for (int c = 0; c < CDIM; c += 4) {
      f32x4 v = *(const f32x4*)(row + c);
      s  += v[0] + v[1] + v[2] + v[3];
      ss += v[0]*v[0] + v[1]*v[1] + v[2]*v[2] + v[3]*v[3];
    }
    float mu = s * (1.0f / CDIM);
    float var = ss * (1.0f / CDIM) - mu * mu;
    float rs = rsqrtf(var + 1e-5f);
    for (int c = 0; c < CDIM; c += 4) {
      f32x4 v = *(const f32x4*)(row + c);
#pragma unroll
      for (int q = 0; q < 4; ++q)
        a_lds[t * CDIM + c + q] = f2bf((v[q] - mu) * rs * ng[c + q] + nb[c + q]);
    }
    m_lds[t] = mask[r0 + t];
  }
  if (wave == 0) __builtin_amdgcn_s_wait_tensorcnt(0);
  __syncthreads();

  // ---- Phase 2: WMMA projections; wave handles two 16-row tiles ----
  for (int st = 0; st < 2; ++st) {
    const int rowbase = wave * 32 + st * 16; // local row base of this tile
    // A fragments (16x32 bf16): lane holds row n16, contiguous K = 16*half + e
    bf16x16 afrag[4];
#pragma unroll
    for (int s = 0; s < 4; ++s) {
      const __bf16* base = a_lds + (rowbase + n16) * CDIM + s * 32 + 16 * half;
      afrag[s] = *(const bf16x16*)base;
    }
    for (int nt = 0; nt < 8; ++nt) {
      f32x8 aL = {}, aR = {}, aLG = {}, aRG = {}, aG = {};
#pragma unroll
      for (int s = 0; s < 4; ++s) {
        size_t boff = (size_t)(nt * 16 + n16) * CDIM + s * 32 + 16 * half;
        bf16x16 bL  = *(const bf16x16*)(w_lds + 0 * CDIM * CDIM + boff);
        bf16x16 bR  = *(const bf16x16*)(w_lds + 1 * CDIM * CDIM + boff);
        bf16x16 bLG = *(const bf16x16*)(w_lds + 2 * CDIM * CDIM + boff);
        bf16x16 bRG = *(const bf16x16*)(w_lds + 3 * CDIM * CDIM + boff);
        bf16x16 bG  = *(const bf16x16*)(w_lds + 4 * CDIM * CDIM + boff);
        aL  = WMMA_BF16(afrag[s], bL,  aL);
        aR  = WMMA_BF16(afrag[s], bR,  aR);
        aLG = WMMA_BF16(afrag[s], bLG, aLG);
        aRG = WMMA_BF16(afrag[s], bRG, aRG);
        aG  = WMMA_BF16(afrag[s], bG,  aG);
      }
      const int cg = nt * 16 + n16; // this lane's output channel
      const float blb = lb[cg], brb = rb[cg], blg = lgb[cg], brg = rgb[cg], bfg = fgb[cg];
      bf16x8 lpack, rpack;
      const size_t growbase = (size_t)r0 + rowbase + 8 * half; // global row of j=0
#pragma unroll
      for (int j = 0; j < 8; ++j) {
        float mk = m_lds[rowbase + 8 * half + j];
        float lv = mk * (aL[j] + blb) * sigmoidf_(aLG[j] + blg);
        float rv = mk * (aR[j] + brb) * sigmoidf_(aRG[j] + brg);
        lpack[j] = f2bf(lv);
        rpack[j] = f2bf(rv);
        gbuf[(growbase + j) * CDIM + cg] = sigmoidf_(aG[j] + bfg);
      }
      // D-fragment lane == one channel across 8 consecutive rows: direct b128 store
      *(bf16x8*)(left_t  + (size_t)cg * NROWS + growbase) = lpack;
      *(bf16x8*)(right_t + (size_t)cg * NROWS + growbase) = rpack;
    }
  }
}

// ---------------------------------------------------------------------------
// Kernel 3: per-channel einsum O[i,j,c] = sum_k L[c][i][k]*R[c][j][k] (WMMA),
// fused LayerNorm over c, fused out-projection (WMMA) + bias + gate multiply.
// Block = one 32x32 (i,j) tile, 512 threads (16 waves, 4/SIMD for latency
// hiding); 2x2 register blocking per wave-channel for L2-traffic reuse.
// Wave w owns channels w*8..w*8+7. LDS o-tile: 32*32*128 bf16 = 256KB.
// ---------------------------------------------------------------------------
__global__ void stage2(const __bf16* __restrict__ left_t, const __bf16* __restrict__ right_t,
                       const float* __restrict__ gbuf,
                       const float* __restrict__ fng, const float* __restrict__ fnb,
                       const float* __restrict__ ob,
                       const __bf16* __restrict__ wt_out,
                       float* __restrict__ out) {
  extern __shared__ char smem[];
  __bf16* o_lds = (__bf16*)smem; // [pos=i_local*32+j_local][c], 1024*128 bf16

  const int t = threadIdx.x;
  const int wave = t >> 5, lane = t & 31;
  const int n16 = lane & 15, half = lane >> 4;
  const int i0 = (blockIdx.x >> 4) * 32;
  const int j0 = (blockIdx.x & 15) * 32;

  // ---- Einsum phase: K = 512, 16 WMMA steps, 2x2 subtile blocking ----
  for (int cc = 0; cc < 8; ++cc) {
    const int c = wave * 8 + cc;
    const __bf16* LbA = left_t  + (size_t)c * NROWS + (size_t)(i0 +      n16) * LDIM;
    const __bf16* LbB = left_t  + (size_t)c * NROWS + (size_t)(i0 + 16 + n16) * LDIM;
    const __bf16* RbA = right_t + (size_t)c * NROWS + (size_t)(j0 +      n16) * LDIM;
    const __bf16* RbB = right_t + (size_t)c * NROWS + (size_t)(j0 + 16 + n16) * LDIM;
    f32x8 acc00 = {}, acc01 = {}, acc10 = {}, acc11 = {};
    for (int ks = 0; ks < 16; ++ks) {
      __builtin_prefetch(LbA + (ks + 2) * 32, 0, 0);
      __builtin_prefetch(RbA + (ks + 2) * 32, 0, 0);
      bf16x16 a0 = *(const bf16x16*)(LbA + ks * 32 + 16 * half);
      bf16x16 a1 = *(const bf16x16*)(LbB + ks * 32 + 16 * half);
      bf16x16 b0 = *(const bf16x16*)(RbA + ks * 32 + 16 * half);
      bf16x16 b1 = *(const bf16x16*)(RbB + ks * 32 + 16 * half);
      acc00 = WMMA_BF16(a0, b0, acc00);
      acc01 = WMMA_BF16(a0, b1, acc01);
      acc10 = WMMA_BF16(a1, b0, acc10);
      acc11 = WMMA_BF16(a1, b1, acc11);
    }
    // D value j of subtile (it,jt): i_local = it*16 + j + 8*half, j_local = jt*16 + n16
#pragma unroll
    for (int j = 0; j < 8; ++j) {
      int ibase0 = (j + 8 * half) * 32;
      int ibase1 = (16 + j + 8 * half) * 32;
      o_lds[(ibase0 +      n16) * CDIM + c] = f2bf(acc00[j]);
      o_lds[(ibase0 + 16 + n16) * CDIM + c] = f2bf(acc01[j]);
      o_lds[(ibase1 +      n16) * CDIM + c] = f2bf(acc10[j]);
      o_lds[(ibase1 + 16 + n16) * CDIM + c] = f2bf(acc11[j]);
    }
  }
  __syncthreads();

  // ---- Fused LayerNorm over c (in place, each thread owns 2 positions) ----
  for (int p = t; p < 1024; p += 512) {
    __bf16* row = o_lds + p * CDIM;
    float s = 0.f, ss = 0.f;
    for (int c = 0; c < CDIM; ++c) { float v = bf2f(row[c]); s += v; ss += v * v; }
    float mu = s * (1.0f / CDIM);
    float var = ss * (1.0f / CDIM) - mu * mu;
    float rs = rsqrtf(var + 1e-5f);
    for (int c = 0; c < CDIM; ++c)
      row[c] = f2bf((bf2f(row[c]) - mu) * rs * fng[c] + fnb[c]);
  }
  __syncthreads();

  // ---- Fused out-projection (WMMA) + bias + sigmoid gate ----
  // 1024 positions / 16 waves = 4 sixteen-row A-tiles per wave.
  for (int st = 0; st < 4; ++st) {
    const int posbase = wave * 64 + st * 16;
    const int i_local = posbase >> 5;        // constant within tile
    const int j_sub   = posbase & 31;        // 0 or 16
    bf16x16 afrag[4];
#pragma unroll
    for (int s = 0; s < 4; ++s) {
      const __bf16* base = o_lds + (posbase + n16) * CDIM + s * 32 + 16 * half;
      afrag[s] = *(const bf16x16*)base;
    }
    const size_t growbase = (size_t)(i0 + i_local) * LDIM + j0 + j_sub + 8 * half; // + j
    for (int nt = 0; nt < 8; ++nt) {
      f32x8 acc = {};
#pragma unroll
      for (int s = 0; s < 4; ++s) {
        bf16x16 b = *(const bf16x16*)(wt_out + (size_t)(nt * 16 + n16) * CDIM + s * 32 + 16 * half);
        acc = WMMA_BF16(afrag[s], b, acc);
      }
      const int cg = nt * 16 + n16;
      const float bias = ob[cg];
#pragma unroll
      for (int j = 0; j < 8; ++j) {
        size_t ridx = (growbase + j) * CDIM + cg;
        out[ridx] = (acc[j] + bias) * gbuf[ridx];
      }
    }
  }
}

// ---------------------------------------------------------------------------
extern "C" void kernel_launch(void* const* d_in, const int* in_sizes, int n_in,
                              void* d_out, int out_size, void* d_ws, size_t ws_size,
                              hipStream_t stream) {
  const float* act   = (const float*)d_in[0];
  const float* pmask = (const float*)d_in[1];
  const float* ng    = (const float*)d_in[2];
  const float* nb    = (const float*)d_in[3];
  const float* lw    = (const float*)d_in[4];
  const float* lb    = (const float*)d_in[5];
  const float* rw    = (const float*)d_in[6];
  const float* rb    = (const float*)d_in[7];
  const float* lgw   = (const float*)d_in[8];
  const float* lgb   = (const float*)d_in[9];
  const float* rgw   = (const float*)d_in[10];
  const float* rgb   = (const float*)d_in[11];
  const float* fng   = (const float*)d_in[12];
  const float* fnb   = (const float*)d_in[13];
  const float* ow    = (const float*)d_in[14];
  const float* obv   = (const float*)d_in[15];
  const float* fgw   = (const float*)d_in[16];
  const float* fgb   = (const float*)d_in[17];
  float* out = (float*)d_out;

  char* ws = (char*)d_ws;
  const size_t LT_BYTES = (size_t)NROWS * CDIM * 2; // 64 MiB
  __bf16* wt      = (__bf16*)ws;                               // 6*16384 bf16
  __bf16* left_t  = (__bf16*)(ws + (1 << 18));                 // 256 KiB offset
  __bf16* right_t = (__bf16*)(ws + (1 << 18) + LT_BYTES);
  float*  gbuf    = (float*) (ws + (1 << 18) + 2 * LT_BYTES);

  pack_weights<<<(6 * CDIM * CDIM + 255) / 256, 256, 0, stream>>>(
      lw, rw, lgw, rgw, fgw, ow, wt);

  const size_t s1_lds = 256 * CDIM * 2 + 256 * 4 + 5 * CDIM * CDIM * 2; // a + mask + weights
  stage1<<<NROWS / 256, 256, s1_lds, stream>>>(
      act, pmask, ng, nb, lb, rb, lgb, rgb, fgb, wt, left_t, right_t, gbuf);

  const size_t s2_lds = 1024 * CDIM * 2; // 256KB o_lds
  stage2<<<(LDIM / 32) * (LDIM / 32), 512, s2_lds, stream>>>(
      left_t, right_t, gbuf, fng, fnb, obv, wt + 5 * CDIM * CDIM, out);
}
